// GNOhMLP_stem_6305011991078
// MI455X (gfx1250) — matrix operand order repaired and checked by reading, not compiled, not run on hardware
//
#include <hip/hip_runtime.h>
#include <hip/hip_bf16.h>
#include <stdint.h>

typedef _Float16 half_t;
typedef __attribute__((ext_vector_type(16))) _Float16 v16h;
typedef __attribute__((ext_vector_type(8)))  float    v8f;

#define NQ    4096
#define NPTS  32768
#define KNBR  32
#define QPB   2
#define ROWS  64          // QPB * KNBR
#define THREADS 128       // 4 waves * 32
#define BATCH 2

// LDS row pitches (in elements), padded +8/+4 to avoid bank conflicts
#define AGG_P 232         // K=224 (196 padded) + 8
#define H0_P  136         // 128 + 8
#define H1_P  264         // 256 + 8
#define FO_P  132         // 128 + 4 (f32)

#define LDS0_BYTES 33792  // max(64*232*2, 64*264*2, 64*132*4)
#define LDS1_BYTES 17408  // 64*136*2

// f16 weight workspace offsets (in halves)
#define W0H_OFF 0         // padded 224 x 128
#define W1H_OFF 28672     // 128 x 256
#define W2H_OFF 61440     // 256 x 128
#define W3H_OFF 94208     // 128 x 128
#define WTOTAL  110592
#define MM_BYTE_OFF (WTOTAL * 2)

__device__ __forceinline__ float gelu_tanh(float v) {
    float v3 = v * v * v;
    return 0.5f * v * (1.0f + tanhf(0.7978845608f * (v + 0.044715f * v3)));
}

// A-fragment (16x32 f16): lane L<16 -> row M=L, K pairs {0..7,16..23};
// lanes 16..31 -> same rows, K pairs shifted by +8.
__device__ __forceinline__ v16h load_a_frag(const half_t* act, int pitch, int rowBase, int kt) {
    const int lane = threadIdx.x & 31;
    const half_t* p = act + (rowBase + (lane & 15)) * pitch + kt * 32 + ((lane >> 4) << 3);
    union { v16h h; uint32_t u[8]; } r;
#pragma unroll
    for (int v = 0; v < 4; ++v) r.u[v]     = *(const uint32_t*)(p + 2 * v);
#pragma unroll
    for (int v = 0; v < 4; ++v) r.u[4 + v] = *(const uint32_t*)(p + 16 + 2 * v);
    return r.h;
}

// One MLP layer: act(ROWSxK f16 in LDS) @ W(KxN f16 in global) + bias, gelu, -> LDS f16
template <int KT, int NT>
__device__ __forceinline__ void mlp_layer(const half_t* __restrict__ actIn, int pitchIn, int rowBase,
                                          const half_t* __restrict__ W, int Ndim,
                                          const float* __restrict__ bias,
                                          half_t* __restrict__ actOut, int pitchOut) {
    const int lane = threadIdx.x & 31;
    const v8f zero = {0.f, 0.f, 0.f, 0.f, 0.f, 0.f, 0.f, 0.f};
    v8f acc[NT];
#pragma unroll
    for (int nt = 0; nt < NT; ++nt) acc[nt] = zero;

#pragma unroll
    for (int kt = 0; kt < KT; ++kt) {
        v16h a = load_a_frag(actIn, pitchIn, rowBase, kt);
        // B-fragment: lane -> column N = lane&15, elements over K in the same order as A
        const half_t* wrow = W + (size_t)(kt * 32 + ((lane >> 4) << 3)) * Ndim + (lane & 15);
#pragma unroll
        for (int nt = 0; nt < NT; ++nt) {
            v16h bf;
#pragma unroll
            for (int s = 0; s < 8; ++s) {
                bf[s]     = wrow[(size_t)s * Ndim + nt * 16];
                bf[8 + s] = wrow[(size_t)(16 + s) * Ndim + nt * 16];
            }
            acc[nt] = __builtin_amdgcn_wmma_f32_16x16x32_f16(
                false, a, false, bf, (short)0, acc[nt], false, false);
        }
    }
    // C/D layout: lane L, reg j -> D[M = j + 8*(L>>4)][N = L&15]
    const int n0 = lane & 15;
    const int mAdd = (lane >> 4) << 3;
#pragma unroll
    for (int nt = 0; nt < NT; ++nt) {
        const float bv = bias[nt * 16 + n0];
#pragma unroll
        for (int j = 0; j < 8; ++j) {
            float g = gelu_tanh(acc[nt][j] + bv);
            actOut[(rowBase + mAdd + j) * pitchOut + nt * 16 + n0] = (half_t)g;
        }
    }
}

// Final layer: no gelu; apply per-row neighbor mask; write f32 to LDS for K-reduction
template <int KT, int NT>
__device__ __forceinline__ void mlp_final(const half_t* __restrict__ actIn, int pitchIn, int rowBase,
                                          const half_t* __restrict__ W, int Ndim,
                                          const float* __restrict__ bias,
                                          const float* __restrict__ maskf,
                                          float* __restrict__ fOut, int pitchOut) {
    const int lane = threadIdx.x & 31;
    const v8f zero = {0.f, 0.f, 0.f, 0.f, 0.f, 0.f, 0.f, 0.f};
    v8f acc[NT];
#pragma unroll
    for (int nt = 0; nt < NT; ++nt) acc[nt] = zero;

#pragma unroll
    for (int kt = 0; kt < KT; ++kt) {
        v16h a = load_a_frag(actIn, pitchIn, rowBase, kt);
        const half_t* wrow = W + (size_t)(kt * 32 + ((lane >> 4) << 3)) * Ndim + (lane & 15);
#pragma unroll
        for (int nt = 0; nt < NT; ++nt) {
            v16h bf;
#pragma unroll
            for (int s = 0; s < 8; ++s) {
                bf[s]     = wrow[(size_t)s * Ndim + nt * 16];
                bf[8 + s] = wrow[(size_t)(16 + s) * Ndim + nt * 16];
            }
            acc[nt] = __builtin_amdgcn_wmma_f32_16x16x32_f16(
                false, a, false, bf, (short)0, acc[nt], false, false);
        }
    }
    const int n0 = lane & 15;
    const int mAdd = (lane >> 4) << 3;
#pragma unroll
    for (int nt = 0; nt < NT; ++nt) {
        const float bv = bias[nt * 16 + n0];
#pragma unroll
        for (int j = 0; j < 8; ++j) {
            const int row = rowBase + mAdd + j;
            fOut[row * pitchOut + nt * 16 + n0] = (acc[nt][j] + bv) * maskf[row];
        }
    }
}

__global__ void __launch_bounds__(THREADS, 1)
gno_mlp_kernel(const float* __restrict__ x,            // (T=1,B,C=4,32,32,32)
               const float* __restrict__ grid_coords,  // (B,32768,3)
               const float* __restrict__ latent_grid,  // (4096,3)
               const int* __restrict__ nbr_idx,        // (B,4096,32)
               const unsigned char* __restrict__ nbr_mask, // bool (B,4096,32)
               const float* __restrict__ b0, const float* __restrict__ b1,
               const float* __restrict__ b2, const float* __restrict__ b3,
               const half_t* __restrict__ w0h, const half_t* __restrict__ w1h,
               const half_t* __restrict__ w2h, const half_t* __restrict__ w3h,
               const float* __restrict__ mm,           // (B,6) bmin/bmax
               float* __restrict__ out) {              // (1,B,128,16,16,16)
    __shared__ __align__(16) unsigned char smem0[LDS0_BYTES];
    __shared__ __align__(16) unsigned char smem1[LDS1_BYTES];
    __shared__ float maskf[ROWS];

    const int b = blockIdx.y;
    const int qBase = blockIdx.x * QPB;
    const int t = threadIdx.x;

    half_t* lds0h = (half_t*)smem0;
    half_t* lds1h = (half_t*)smem1;
    float*  lds0f = (float*)smem0;

    if (t < ROWS) {
        const int q = qBase + (t >> 5);
        maskf[t] = nbr_mask[(size_t)(b * NQ + q) * KNBR + (t & 31)] ? 1.0f : 0.0f;
    }

    // -------- build agg features (ROWS x 224 f16 in lds0), 2 threads per row
    {
        const int r = t >> 1;
        const int role = t & 1;
        const int q = qBase + (r >> 5);
        const int k = r & 31;
        const int n = nbr_idx[(size_t)(b * NQ + q) * KNBR + k];
        half_t* row = lds0h + r * AGG_P;
        const float LNF = 0.575646273f;  // ln(10000)/16
        if (role == 0) {
            // neighbor-coord embedding (cols 0..95) + raw features (192..195)
            const float* gy = grid_coords + ((size_t)b * NPTS + n) * 3;
#pragma unroll
            for (int c0 = 0; c0 < 3; ++c0) {
                const float coord = gy[c0];
                for (int j = 0; j < 16; ++j) {
                    const float ang = coord * __expf(-(float)j * LNF);
                    row[c0 * 32 + j]      = (half_t)__sinf(ang);
                    row[c0 * 32 + 16 + j] = (half_t)__cosf(ang);
                }
            }
            // f[b,0,n,c] = x[0,b,c,d,h,w] with n = (h*32+w)*32+d
            const int d = n & 31, w = (n >> 5) & 31, h = n >> 10;
            const int sp = (d << 10) + (h << 5) + w;
#pragma unroll
            for (int c = 0; c < 4; ++c)
                row[192 + c] = (half_t)x[(((size_t)b * 4 + c) << 15) + sp];
        } else {
            // latent-coord embedding (cols 96..191) + zero K-pad (196..223)
            const float* mmb = mm + b * 6;
            const float* lg = latent_grid + (size_t)q * 3;
#pragma unroll
            for (int c0 = 0; c0 < 3; ++c0) {
                const float coord = mmb[c0] + (mmb[3 + c0] - mmb[c0]) * lg[c0];
                for (int j = 0; j < 16; ++j) {
                    const float ang = coord * __expf(-(float)j * LNF);
                    row[96 + c0 * 32 + j]      = (half_t)__sinf(ang);
                    row[96 + c0 * 32 + 16 + j] = (half_t)__cosf(ang);
                }
            }
            for (int c = 196; c < 224; ++c) row[c] = (half_t)0.f;
        }
    }
    __syncthreads();

    // -------- fused 4-layer MLP, one 16-row WMMA strip per wave
    const int rowBase = (t >> 5) * 16;
    mlp_layer<7, 8>(lds0h, AGG_P, rowBase, w0h, 128, b0, lds1h, H0_P);   // 224->128
    __syncthreads();
    mlp_layer<4, 16>(lds1h, H0_P, rowBase, w1h, 256, b1, lds0h, H1_P);   // 128->256
    __syncthreads();
    mlp_layer<8, 8>(lds0h, H1_P, rowBase, w2h, 128, b2, lds1h, H0_P);    // 256->128
    __syncthreads();
    mlp_final<4, 8>(lds1h, H0_P, rowBase, w3h, 128, b3, maskf, lds0f, FO_P); // 128->128
    __syncthreads();

    // -------- masked sum over K=32 neighbors, write transposed output
    for (int p = t; p < QPB * 128; p += THREADS) {
        const int q_local = p >> 7;
        const int ch = p & 127;
        float s = 0.f;
#pragma unroll 8
        for (int k = 0; k < KNBR; ++k)
            s += lds0f[(q_local * KNBR + k) * FO_P + ch];
        const int q = qBase + q_local;
        const int dl = q & 15, wl = (q >> 4) & 15, hl = q >> 8;
        // out[t=0, b, ch, dl, hl, wl]
        out[(((size_t)b * 128 + ch) << 12) + dl * 256 + hl * 16 + wl] = s;
    }
}

// per-batch coord min/max over 32768 points
__global__ void minmax_kernel(const float* __restrict__ gc, float* __restrict__ mm) {
    const int b = blockIdx.x;
    const float* g = gc + (size_t)b * NPTS * 3;
    float mn[3] = {1e30f, 1e30f, 1e30f}, mx[3] = {-1e30f, -1e30f, -1e30f};
    for (int i = threadIdx.x; i < NPTS; i += blockDim.x) {
#pragma unroll
        for (int c = 0; c < 3; ++c) {
            const float v = g[i * 3 + c];
            mn[c] = fminf(mn[c], v);
            mx[c] = fmaxf(mx[c], v);
        }
    }
    __shared__ float s[6][256];
    const int t = threadIdx.x;
#pragma unroll
    for (int c = 0; c < 3; ++c) { s[c][t] = mn[c]; s[3 + c][t] = mx[c]; }
    __syncthreads();
    for (int off = 128; off > 0; off >>= 1) {
        if (t < off) {
#pragma unroll
            for (int c = 0; c < 3; ++c) {
                s[c][t]     = fminf(s[c][t], s[c][t + off]);
                s[3 + c][t] = fmaxf(s[3 + c][t], s[3 + c][t + off]);
            }
        }
        __syncthreads();
    }
    if (t == 0) {
#pragma unroll
        for (int c = 0; c < 6; ++c) mm[b * 6 + c] = s[c][0];
    }
}

// f32 -> f16 weights, W0 zero-padded from K=196 to K=224
__global__ void convert_weights_kernel(const float* __restrict__ W0, const float* __restrict__ W1,
                                       const float* __restrict__ W2, const float* __restrict__ W3,
                                       half_t* __restrict__ wsH) {
    const int i = blockIdx.x * blockDim.x + threadIdx.x;
    if (i >= WTOTAL) return;
    float v;
    if (i < W1H_OFF) {
        const int k = i >> 7, n = i & 127;
        v = (k < 196) ? W0[k * 128 + n] : 0.0f;
    } else if (i < W2H_OFF) {
        v = W1[i - W1H_OFF];
    } else if (i < W3H_OFF) {
        v = W2[i - W2H_OFF];
    } else {
        v = W3[i - W3H_OFF];
    }
    wsH[i] = (half_t)v;
}

extern "C" void kernel_launch(void* const* d_in, const int* in_sizes, int n_in,
                              void* d_out, int out_size, void* d_ws, size_t ws_size,
                              hipStream_t stream) {
    const float* x   = (const float*)d_in[0];
    const float* gc  = (const float*)d_in[1];
    const float* lg  = (const float*)d_in[2];
    const int*   idx = (const int*)d_in[3];
    const unsigned char* msk = (const unsigned char*)d_in[4];
    const float* W0 = (const float*)d_in[5];
    const float* b0 = (const float*)d_in[6];
    const float* W1 = (const float*)d_in[7];
    const float* b1 = (const float*)d_in[8];
    const float* W2 = (const float*)d_in[9];
    const float* b2 = (const float*)d_in[10];
    const float* W3 = (const float*)d_in[11];
    const float* b3 = (const float*)d_in[12];
    float* out = (float*)d_out;

    half_t* wsH = (half_t*)d_ws;
    float* mm = (float*)((char*)d_ws + MM_BYTE_OFF);

    convert_weights_kernel<<<(WTOTAL + 255) / 256, 256, 0, stream>>>(W0, W1, W2, W3, wsH);
    minmax_kernel<<<BATCH, 256, 0, stream>>>(gc, mm);
    gno_mlp_kernel<<<dim3(NQ / QPB, BATCH), THREADS, 0, stream>>>(
        x, gc, lg, idx, msk, b0, b1, b2, b3,
        wsH + W0H_OFF, wsH + W1H_OFF, wsH + W2H_OFF, wsH + W3H_OFF,
        mm, out);
}
